// Pooling_22754736735005
// MI455X (gfx1250) — matrix operand order repaired and checked by reading, not compiled, not run on hardware
//
#include <hip/hip_runtime.h>
#include <hip/hip_bf16.h>

#define GN 4096
#define GH 512
#define BM 256    // block M tile (8 waves x 32 rows)
#define BN 64     // block N tile (4 x 16)
#define BK 64     // K step (2 WMMA K-slabs)
#define ASTR 72   // padded LDS stride in halves: 144B = 16B-aligned, 36-bank step (conflict free)

typedef __attribute__((ext_vector_type(16))) _Float16 v16h;
typedef __attribute__((ext_vector_type(8)))  _Float16 v8h;
typedef __attribute__((ext_vector_type(8)))  float    v8f;

// ---------------------------------------------------------------------------
// GEMM: D[m,n] = sum_k Aop[m,k] * Bt[n,k]   (i.e. D = Aop @ Bt^T), all 4096^2.
// MODE 0: write f32 D (final next_adj)
// MODE 1: write f16 reach mask: ((acc>0 || Aorig[m,n]>0) && m!=n) ? 1 : 0
// MODE 2: write f16 D (exact small integer counts)
// Each wave computes a 32x64 output tile = 2(M) x 4(N) WMMA tiles.
// ---------------------------------------------------------------------------
template <int MODE>
__global__ __launch_bounds__(256) void gemm16(const _Float16* __restrict__ Aop,
                                              const _Float16* __restrict__ Bt,
                                              float* __restrict__ Df,
                                              _Float16* __restrict__ Dh,
                                              const _Float16* __restrict__ Aorig) {
  __shared__ _Float16 As[BM * ASTR];   // 256 x 64 (padded)
  __shared__ _Float16 Bs[BN * ASTR];   // 64 x 64 (padded)

  const int tid  = threadIdx.x;
  const int lane = tid & 31;
  const int w    = tid >> 5;        // wave 0..7 -> 32-row M sub-tile
  const int l15  = lane & 15;
  const int hi   = lane >> 4;       // 0: lanes 0-15, 1: lanes 16-31
  const int m0   = blockIdx.y * BM;
  const int n0   = blockIdx.x * BN;

  v8f acc[2][4];
#pragma unroll
  for (int u = 0; u < 2; ++u)
#pragma unroll
    for (int t = 0; t < 4; ++t) acc[u][t] = (v8f)0.0f;

  for (int kt = 0; kt < GN; kt += BK) {
    __syncthreads();
    // ---- stage: gather all global chunks to registers first (overlapped
    //      loads), then commit to LDS ----
    v8h ar[8];   // A tile: 256x64 halves = 2048 chunks of 8 / 256 threads
    v8h br[2];   // Bt tile: 64x64 halves = 512 chunks of 8 / 256 threads
#pragma unroll
    for (int it = 0; it < 8; ++it) {
      int idx = tid + it * 256;
      int r = idx >> 3;
      int c = (idx & 7) * 8;
      ar[it] = *(const v8h*)&Aop[(size_t)(m0 + r) * GN + kt + c];
    }
#pragma unroll
    for (int it = 0; it < 2; ++it) {
      int idx = tid + it * 256;
      int r = idx >> 3;
      int c = (idx & 7) * 8;
      br[it] = *(const v8h*)&Bt[(size_t)(n0 + r) * GN + kt + c];
    }
#pragma unroll
    for (int it = 0; it < 8; ++it) {
      int idx = tid + it * 256;
      int r = idx >> 3;
      int c = (idx & 7) * 8;
      *(v8h*)&As[r * ASTR + c] = ar[it];
    }
#pragma unroll
    for (int it = 0; it < 2; ++it) {
      int idx = tid + it * 256;
      int r = idx >> 3;
      int c = (idx & 7) * 8;
      *(v8h*)&Bs[r * ASTR + c] = br[it];
    }
    __syncthreads();

#pragma unroll
    for (int kk = 0; kk < BK; kk += 32) {
      // A fragments (16x32 f16): lane holds row, K = {kk+8*hi..+7} U {kk+16+8*hi..+7}
      v16h a[2];
#pragma unroll
      for (int u = 0; u < 2; ++u) {
        const _Float16* ap = &As[(w * 32 + u * 16 + l15) * ASTR + kk + hi * 8];
        v8h alo = *(const v8h*)ap;
        v8h ahi = *(const v8h*)(ap + 16);
        a[u] = __builtin_shufflevector(alo, ahi, 0, 1, 2, 3, 4, 5, 6, 7,
                                       8, 9, 10, 11, 12, 13, 14, 15);
      }
      // B fragments (32x16 f16): lane holds column, K = {kk+8*hi..+7} U {kk+16+8*hi..+7}
      v16h b[4];
#pragma unroll
      for (int t = 0; t < 4; ++t) {
        const _Float16* bp = &Bs[(t * 16 + l15) * ASTR + kk + hi * 8];
        v8h blo = *(const v8h*)bp;
        v8h bhi = *(const v8h*)(bp + 16);
        b[t] = __builtin_shufflevector(blo, bhi, 0, 1, 2, 3, 4, 5, 6, 7,
                                       8, 9, 10, 11, 12, 13, 14, 15);
      }
#pragma unroll
      for (int u = 0; u < 2; ++u)
#pragma unroll
        for (int t = 0; t < 4; ++t)
          acc[u][t] = __builtin_amdgcn_wmma_f32_16x16x32_f16(
              false, a[u], false, b[t], (short)0, acc[u][t], false, false);
    }
  }

  // epilogue: C/D layout: VGPR v, lanes 0-15 -> M = v, lanes 16-31 -> M = v+8; N = l15
#pragma unroll
  for (int u = 0; u < 2; ++u) {
#pragma unroll
    for (int t = 0; t < 4; ++t) {
      int col = n0 + t * 16 + l15;
#pragma unroll
      for (int v = 0; v < 8; ++v) {
        int row  = m0 + w * 32 + u * 16 + hi * 8 + v;
        size_t o = (size_t)row * GN + col;
        float val = acc[u][t][v];
        if (MODE == 0) {
          Df[o] = val;
        } else if (MODE == 1) {
          // branchless: avoid exec-mask divergence around the Aorig load
          float aorig = (float)Aorig[o];
          int r = ((int)(val > 0.5f) | (int)(aorig > 0.5f)) & (int)(row != col);
          Dh[o] = (_Float16)(float)r;
        } else {
          Dh[o] = (_Float16)val;   // exact: integer counts << 2048
        }
      }
    }
  }
}

// ---------------------------------------------------------------------------
// 64x64 f16 transpose (LDS tiled)
// ---------------------------------------------------------------------------
__global__ __launch_bounds__(256) void transpose_h(const _Float16* __restrict__ in,
                                                   _Float16* __restrict__ out) {
  __shared__ _Float16 tile[64][65];
  int tx = threadIdx.x & 63;
  int ty = threadIdx.x >> 6;  // 0..3
  int bx = blockIdx.x, by = blockIdx.y;
#pragma unroll
  for (int r = ty; r < 64; r += 4)
    tile[r][tx] = in[(size_t)(by * 64 + r) * GN + bx * 64 + tx];
  __syncthreads();
#pragma unroll
  for (int r = ty; r < 64; r += 4)
    out[(size_t)(bx * 64 + r) * GN + by * 64 + tx] = tile[tx][r];
}

__global__ __launch_bounds__(256) void zero_u32(unsigned* __restrict__ p, int n) {
  int i = blockIdx.x * 256 + threadIdx.x;
  if (i < n) p[i] = 0u;
}

__global__ __launch_bounds__(256) void scatter_edges(const int* __restrict__ ei, int ne,
                                                     _Float16* __restrict__ Ah) {
  int e = blockIdx.x * 256 + threadIdx.x;
  if (e >= ne) return;
  int s = ei[e], d = ei[ne + e];
  Ah[(size_t)s * GN + d] = (_Float16)1.0f;   // duplicates write same value
}

// h[i] = x[i,:] . W  (one wave per row)
__global__ __launch_bounds__(256) void score_h(const float* __restrict__ x,
                                               const float* __restrict__ W,
                                               float* __restrict__ h) {
  int i = blockIdx.x * 8 + (threadIdx.x >> 5);
  int lane = threadIdx.x & 31;
  float s = 0.0f;
  for (int k = lane; k < GH; k += 32) s += x[(size_t)i * GH + k] * W[k];
#pragma unroll
  for (int m = 16; m >= 1; m >>= 1) s += __shfl_xor(s, m, 32);
  if (lane == 0) h[i] = s;
}

// per-column: indeg[j] = sum_i M[i,j]; fitness[j] = sigmoid(h[j] + (h.M)[j]/max(indeg,1))
__global__ __launch_bounds__(256) void col_stats(const _Float16* __restrict__ M,
                                                 const float* __restrict__ h,
                                                 float* __restrict__ indeg,
                                                 float* __restrict__ fit) {
  int j = blockIdx.x * 256 + threadIdx.x;
  float id = 0.0f, hm = 0.0f;
  for (int i = 0; i < GN; ++i) {
    float m = (float)M[(size_t)i * GN + j];
    id += m;
    hm = fmaf(h[i], m, hm);
  }
  float agg = hm / fmaxf(id, 1.0f);
  indeg[j] = id;
  float z = h[j] + agg;
  fit[j] = 1.0f / (1.0f + expf(-z));
}

__global__ __launch_bounds__(256) void edge_masks(const int* __restrict__ ei, int ne,
                                                  const float* __restrict__ fit,
                                                  int* __restrict__ allm,
                                                  int* __restrict__ delm) {
  int e = blockIdx.x * 256 + threadIdx.x;
  if (e >= ne) return;
  int s = ei[e], d = ei[ne + e];
  if (s != d) {
    atomicOr(&allm[s], 1);
    atomicOr(&allm[d], 1);
    if (fit[d] - fit[s] < 0.0f) atomicOr(&delm[d], 1);
  }
}

__global__ __launch_bounds__(256) void select_k(const int* __restrict__ allm,
                                                const int* __restrict__ delm,
                                                int* __restrict__ sel) {
  int i = blockIdx.x * 256 + threadIdx.x;
  sel[i] = (allm[i] != 0) && (delm[i] == 0);
}

// rowany[i] = OR_j (M[i,j] && sel[j])  (one wave per row, coalesced)
__global__ __launch_bounds__(256) void rowany_k(const _Float16* __restrict__ M,
                                                const int* __restrict__ sel,
                                                int* __restrict__ rowany) {
  int i = blockIdx.x * 8 + (threadIdx.x >> 5);
  int lane = threadIdx.x & 31;
  int f = 0;
  for (int j = lane; j < GN; j += 32)
    f |= ((int)((float)M[(size_t)i * GN + j] > 0.5f)) & (int)(sel[j] != 0);
#pragma unroll
  for (int m = 16; m >= 1; m >>= 1) f |= __shfl_xor(f, m, 32);
  if (lane == 0) rowany[i] = f;
}

__global__ __launch_bounds__(256) void nextmask_k(const int* __restrict__ allm,
                                                  const int* __restrict__ sel,
                                                  const int* __restrict__ rowany,
                                                  const float* __restrict__ indeg,
                                                  const int* __restrict__ batch,
                                                  int* __restrict__ nextm,
                                                  float* __restrict__ nb_out) {
  int i = blockIdx.x * 256 + threadIdx.x;
  int colany = (sel[i] != 0) && (indeg[i] > 0.5f);     // Msel.any(axis=0)
  int deln = colany || rowany[i];                       // | Msel.any(axis=1)
  int surplus = (allm[i] != 0) && !deln;
  int nm = (sel[i] != 0) || surplus;
  nextm[i] = nm;
  nb_out[i] = nm ? (float)batch[i] : -1.0f;
}

// S[i,j] = (M[i,j] && sel[j]) + (i==j && nextm[i]); write f16 (for GEMM) and f32 (output)
__global__ __launch_bounds__(256) void build_S(const _Float16* __restrict__ M,
                                               const int* __restrict__ sel,
                                               const int* __restrict__ nextm,
                                               _Float16* __restrict__ Sh,
                                               float* __restrict__ Sout) {
  int j = blockIdx.x * 256 + threadIdx.x;
  int i = blockIdx.y;
  float s = (((float)M[(size_t)i * GN + j] > 0.5f) && (sel[j] != 0)) ? 1.0f : 0.0f;
  if (i == j && nextm[i]) s = 1.0f;                     // diag of Msel is 0, never 2.0
  size_t o = (size_t)i * GN + j;
  Sh[o] = (_Float16)s;
  Sout[o] = s;
}

__global__ __launch_bounds__(256) void next_x_k(const float* __restrict__ x,
                                                const float* __restrict__ fit,
                                                const int* __restrict__ nextm,
                                                float* __restrict__ out) {
  int k = blockIdx.x * 256 + threadIdx.x;  // 0..511
  int i = blockIdx.y;
  float f = nextm[i] ? fit[i] : 0.0f;
  out[(size_t)i * GH + k] = x[(size_t)i * GH + k] * f;
}

// ---------------------------------------------------------------------------
extern "C" void kernel_launch(void* const* d_in, const int* in_sizes, int n_in,
                              void* d_out, int out_size, void* d_ws, size_t ws_size,
                              hipStream_t stream) {
  const float* x     = (const float*)d_in[0];
  const int*   ei    = (const int*)d_in[1];
  const int*   batch = (const int*)d_in[2];
  const float* Wsc   = (const float*)d_in[3];
  const int ne = in_sizes[1] / 2;  // 131072

  float* out     = (float*)d_out;
  float* out_nx  = out;                                  // [GN*GH]
  float* out_adj = out + (size_t)GN * GH;                // [GN*GN]
  float* out_S   = out_adj + (size_t)GN * GN;            // [GN*GN]
  float* out_nb  = out_S + (size_t)GN * GN;              // [GN]

  // Scratch: the 64MB next_adj output region hosts two f16 4096^2 buffers
  // (both dead before the final GEMM overwrites the region).
  _Float16* Ah = (_Float16*)out_adj;                     // adjacency (f16)
  _Float16* Mh = Ah + (size_t)GN * GN;                   // reach mask M, later T = A@S

  char* ws = (char*)d_ws;
  const size_t MATB = (size_t)GN * GN * sizeof(_Float16);  // 32MB
  _Float16* ATh = (_Float16*)ws;                         // A^T, later T^T
  _Float16* Sh  = (_Float16*)(ws + MATB);                // S (f16)
  _Float16* STh = (_Float16*)(ws + 2 * MATB);            // S^T (f16)
  float* hvec   = (float*)(ws + 3 * MATB);
  float* indeg  = hvec + GN;
  float* fit    = indeg + GN;
  int* allm     = (int*)(fit + GN);
  int* delm     = allm + GN;
  int* sel      = delm + GN;
  int* rowany   = sel + GN;
  int* nextm    = rowany + GN;

  const dim3 blk(256);
  const dim3 gemm_grid(GN / BN, GN / BM);   // 64 x 16
  const dim3 tr_grid(GN / 64, GN / 64);     // 64 x 64

  // 1) A = scatter(edges), A^T
  zero_u32<<<dim3((GN * (GN / 2) + 255) / 256), blk, 0, stream>>>((unsigned*)Ah, GN * (GN / 2));
  zero_u32<<<dim3((2 * GN + 255) / 256), blk, 0, stream>>>((unsigned*)allm, 2 * GN);
  scatter_edges<<<dim3((ne + 255) / 256), blk, 0, stream>>>(ei, ne, Ah);
  transpose_h<<<tr_grid, blk, 0, stream>>>(Ah, ATh);

  // 2) M = ((A@A > 0) | (A > 0)) & ~I     [WMMA GEMM 1]
  gemm16<1><<<gemm_grid, blk, 0, stream>>>(Ah, ATh, (float*)nullptr, Mh, Ah);

  // 3) scores / fitness / node masks
  score_h<<<dim3(GN / 8), blk, 0, stream>>>(x, Wsc, hvec);
  col_stats<<<dim3(GN / 256), blk, 0, stream>>>(Mh, hvec, indeg, fit);
  edge_masks<<<dim3((ne + 255) / 256), blk, 0, stream>>>(ei, ne, fit, allm, delm);
  select_k<<<dim3(GN / 256), blk, 0, stream>>>(allm, delm, sel);
  rowany_k<<<dim3(GN / 8), blk, 0, stream>>>(Mh, sel, rowany);
  nextmask_k<<<dim3(GN / 256), blk, 0, stream>>>(allm, sel, rowany, indeg, batch, nextm, out_nb);

  // 4) S (f16 + f32 output), S^T, next_x
  build_S<<<dim3(GN / 256, GN), blk, 0, stream>>>(Mh, sel, nextm, Sh, out_S);
  transpose_h<<<tr_grid, blk, 0, stream>>>(Sh, STh);
  next_x_k<<<dim3(GH / 256, GN), blk, 0, stream>>>(x, fit, nextm, out_nx);

  // 5) T = A @ S (f16 exact counts, reuses M's buffer)   [WMMA GEMM 2]
  gemm16<2><<<gemm_grid, blk, 0, stream>>>(Ah, STh, (float*)nullptr, Mh, (const _Float16*)nullptr);
  // 6) T^T (reuses A^T buffer)
  transpose_h<<<tr_grid, blk, 0, stream>>>(Mh, ATh);
  // 7) next_adj = S^T @ T -> f32 output (overwrites Ah/Mh scratch region)  [WMMA GEMM 3]
  gemm16<0><<<gemm_grid, blk, 0, stream>>>(STh, ATh, out_adj, (_Float16*)nullptr,
                                           (const _Float16*)nullptr);
}
